// MambaBlock_58067957842028
// MI455X (gfx1250) — matrix operand order repaired and checked by reading, not compiled, not run on hardware
//
#include <hip/hip_runtime.h>
#include <hip/hip_bf16.h>
#include <stddef.h>

// ---------------- problem constants ----------------
#define D_MODEL 1024
#define D_STATE 16
#define D_CONV  4
#define D_INNER 2048
#define B_SZ    2
#define T_LEN   2048
#define NROWS   (B_SZ * T_LEN)          // 4096 token rows
#define N_XZ    (2 * D_INNER)           // 4096
#define EPSF    1e-6f

typedef __attribute__((ext_vector_type(16))) _Float16 v16h;
typedef __attribute__((ext_vector_type(8)))  _Float16 v8h;
typedef __attribute__((ext_vector_type(8)))  float    v8f;

// ------- gfx1250 async global->LDS path (guarded, with sync fallback) -------
#if defined(__AMDGCN__) && __has_builtin(__builtin_amdgcn_global_load_async_to_lds_b128)
#define HAVE_ASYNC_LDS 1
#else
#define HAVE_ASYNC_LDS 0
#endif

#if HAVE_ASYNC_LDS
// builtin parameter types (from clang diagnostic):
//   arg0: int __vector(4) __device__ * (addrspace(1))
//   arg1: LDS pointer (addrspace(3))
typedef int v4i_t __attribute__((vector_size(16)));
typedef __attribute__((address_space(1))) v4i_t* gptr_v4i;
typedef __attribute__((address_space(3))) v4i_t* lptr_v4i;

__device__ __forceinline__ void async_copy_b128(const void* gsrc, void* ldst) {
    __builtin_amdgcn_global_load_async_to_lds_b128(
        (gptr_v4i)(v4i_t*)(void*)gsrc,
        (lptr_v4i)(v4i_t*)ldst,
        /*imm offset=*/0, /*cpol=*/0);
}
__device__ __forceinline__ void wait_async0() {
#if __has_builtin(__builtin_amdgcn_s_wait_asynccnt)
    __builtin_amdgcn_s_wait_asynccnt(0);
#else
    asm volatile("s_wait_asynccnt 0x0" ::: "memory");
#endif
}
#endif

// ---------------- 1) RMSNorm + cast to f16 ----------------
__global__ __launch_bounds__(256) void rmsnorm_cast_kernel(
    const float* __restrict__ x, const float* __restrict__ w,
    _Float16* __restrict__ out)
{
    const int row = blockIdx.x;                  // 0..NROWS-1
    const float* xr = x + (size_t)row * D_MODEL;
    float ss = 0.f;
    for (int i = threadIdx.x; i < D_MODEL; i += 256) {
        float v = xr[i];
        ss += v * v;
    }
    __shared__ float red[256];
    red[threadIdx.x] = ss;
    __syncthreads();
    for (int s = 128; s > 0; s >>= 1) {
        if (threadIdx.x < s) red[threadIdx.x] += red[threadIdx.x + s];
        __syncthreads();
    }
    const float rinv = rsqrtf(red[0] * (1.0f / D_MODEL) + EPSF);
    for (int i = threadIdx.x; i < D_MODEL; i += 256)
        out[(size_t)row * D_MODEL + i] = (_Float16)(xr[i] * rinv * w[i]);
}

// ---------------- 2) transpose + cast f32 -> f16 ----------------
// in: rows x cols (row-major).  out: cols x rows (row-major) i.e. in^T.
__global__ __launch_bounds__(256) void transpose_cast_kernel(
    const float* __restrict__ in, _Float16* __restrict__ out,
    int rows, int cols)
{
    __shared__ float tile[32][33];
    const int r0 = blockIdx.y * 32;
    const int c0 = blockIdx.x * 32;
    const int tx = threadIdx.x & 31;
    const int ty = threadIdx.x >> 5;             // 0..7
    for (int i = ty; i < 32; i += 8)
        tile[i][tx] = in[(size_t)(r0 + i) * cols + (c0 + tx)];
    __syncthreads();
    for (int i = ty; i < 32; i += 8)
        out[(size_t)(c0 + i) * rows + (r0 + tx)] = (_Float16)tile[tx][i];
}

// ---------------- 3) WMMA GEMM: C[M,N] = A[M,K] * Bt[N,K]^T (+ residual) ----
// Block tile 128x128x32, 8 waves (4x2), each wave 32x64 = 2x4 WMMA tiles.
// Double-buffered LDS; global->LDS via async DMA (ASYNCcnt) when available.
#define BM 128
#define BN 128
#define BK 32
#define LDS_K (BK + 16)   // 48 halves => 96B row stride, keeps 16B alignment

template <bool ADD_RES>
__global__ __launch_bounds__(256) void gemm_wmma_f16(
    const _Float16* __restrict__ A,   // M x K  row-major (f16)
    const _Float16* __restrict__ Bt,  // N x K  row-major (f16) == B^T
    float* __restrict__ C,            // M x N  row-major (f32)
    const float* __restrict__ Res,    // M x N  or nullptr
    int M, int N, int K)
{
    __shared__ __align__(16) _Float16 sA[2][BM][LDS_K];
    __shared__ __align__(16) _Float16 sB[2][BN][LDS_K];

    const int bm   = blockIdx.y * BM;
    const int bn   = blockIdx.x * BN;
    const int tid  = threadIdx.x;
    const int wave = tid >> 5;
    const int lane = tid & 31;
    const int wm   = wave >> 1;                 // 0..3  -> 32-row slab
    const int wn   = wave & 1;                  // 0..1  -> 64-col slab
    const int hlf  = lane >> 4;                 // lane half (0/1)
    const int l16  = lane & 15;

    // per-thread tile-fill coordinates: rows {r, r+64}, one 16B chunk each
    const int lrow = tid >> 2;                  // 0..63
    const int lcol = (tid & 3) * 8;             // 0,8,16,24

    v8f acc[2][4];
    #pragma unroll
    for (int mi = 0; mi < 2; ++mi)
        #pragma unroll
        for (int ni = 0; ni < 4; ++ni)
            acc[mi][ni] = (v8f){0.f,0.f,0.f,0.f,0.f,0.f,0.f,0.f};

    const int nkt = K / BK;

    // ---- prologue: stage k-tile 0 into buffer 0 ----
    {
#if HAVE_ASYNC_LDS
        #pragma unroll
        for (int i = 0; i < 2; ++i) {
            const int r = lrow + i * 64;
            async_copy_b128(&A [(size_t)(bm + r) * K + lcol], &sA[0][r][lcol]);
            async_copy_b128(&Bt[(size_t)(bn + r) * K + lcol], &sB[0][r][lcol]);
        }
        wait_async0();
#else
        uint4 ra[2], rb[2];
        #pragma unroll
        for (int i = 0; i < 2; ++i) {
            const int r = lrow + i * 64;
            ra[i] = *(const uint4*)&A [(size_t)(bm + r) * K + lcol];
            rb[i] = *(const uint4*)&Bt[(size_t)(bn + r) * K + lcol];
        }
        #pragma unroll
        for (int i = 0; i < 2; ++i) {
            const int r = lrow + i * 64;
            *(uint4*)&sA[0][r][lcol] = ra[i];
            *(uint4*)&sB[0][r][lcol] = rb[i];
        }
#endif
    }
    __syncthreads();

    for (int kt = 0; kt < nkt; ++kt) {
        const int cur = kt & 1;
        const int nxt = cur ^ 1;
        const bool have_next = (kt + 1) < nkt;
        const int k0n = (kt + 1) * BK;

#if !HAVE_ASYNC_LDS
        uint4 ra[2], rb[2];
#endif
        if (have_next) {
            // stream next K tile while this one is consumed
            #pragma unroll
            for (int i = 0; i < 2; ++i) {
                const int r = lrow + i * 64;
#if HAVE_ASYNC_LDS
                async_copy_b128(&A [(size_t)(bm + r) * K + k0n + lcol],
                                &sA[nxt][r][lcol]);
                async_copy_b128(&Bt[(size_t)(bn + r) * K + k0n + lcol],
                                &sB[nxt][r][lcol]);
#else
                ra[i] = *(const uint4*)&A [(size_t)(bm + r) * K + k0n + lcol];
                rb[i] = *(const uint4*)&Bt[(size_t)(bn + r) * K + k0n + lcol];
#endif
            }
        }

        // A fragments (ISA 16-bit A 16x32 layout):
        //   e=0..7  -> K = hlf*8 + e ;  e=8..15 -> K = 16 + hlf*8 + (e-8)
        v16h afrag[2];
        #pragma unroll
        for (int mi = 0; mi < 2; ++mi) {
            const int m = wm * 32 + mi * 16 + l16;
            v8h alo = *(const v8h*)&sA[cur][m][hlf * 8];
            v8h ahi = *(const v8h*)&sA[cur][m][16 + hlf * 8];
            afrag[mi] = __builtin_shufflevector(alo, ahi,
                0,1,2,3,4,5,6,7,8,9,10,11,12,13,14,15);
        }
        // B fragments: e=0..7 -> K = hlf*8 + e ; e=8..15 -> K = 16 + hlf*8 + (e-8)
        v16h bfrag[4];
        #pragma unroll
        for (int ni = 0; ni < 4; ++ni) {
            const int n = wn * 64 + ni * 16 + l16;
            v8h blo = *(const v8h*)&sB[cur][n][hlf * 8];
            v8h bhi = *(const v8h*)&sB[cur][n][16 + hlf * 8];
            bfrag[ni] = __builtin_shufflevector(blo, bhi,
                0,1,2,3,4,5,6,7,8,9,10,11,12,13,14,15);
        }

        #pragma unroll
        for (int mi = 0; mi < 2; ++mi)
            #pragma unroll
            for (int ni = 0; ni < 4; ++ni)
                acc[mi][ni] = __builtin_amdgcn_wmma_f32_16x16x32_f16(
                    /*neg_a=*/false, afrag[mi],
                    /*neg_b=*/false, bfrag[ni],
                    /*c_mod=*/(short)0, acc[mi][ni],
                    /*reuse_a=*/false, /*reuse_b=*/false);

        if (have_next) {
#if HAVE_ASYNC_LDS
            wait_async0();          // this wave's DMA into nxt buffer done
#else
            #pragma unroll
            for (int i = 0; i < 2; ++i) {
                const int r = lrow + i * 64;
                *(uint4*)&sA[nxt][r][lcol] = ra[i];
                *(uint4*)&sB[nxt][r][lcol] = rb[i];
            }
#endif
        }
        __syncthreads();            // publish nxt, retire cur
    }

    // writeback: VGPR r of C/D holds row M = r + 8*hlf, col N = l16
    #pragma unroll
    for (int mi = 0; mi < 2; ++mi) {
        const int rbase = bm + wm * 32 + mi * 16 + hlf * 8;
        #pragma unroll
        for (int ni = 0; ni < 4; ++ni) {
            const int col = bn + wn * 64 + ni * 16 + l16;
            #pragma unroll
            for (int r = 0; r < 8; ++r) {
                const size_t off = (size_t)(rbase + r) * N + col;
                float v = acc[mi][ni][r];
                if (ADD_RES) v += Res[off];
                C[off] = v;
            }
        }
    }
}

// ---------------- 4) causal depthwise conv (k=4) + SiLU ----------------
__global__ __launch_bounds__(256) void conv_silu_kernel(
    const float* __restrict__ xz,     // NROWS x N_XZ ; x_proj = cols [0,D_INNER)
    const float* __restrict__ cw,     // D_INNER x 4
    const float* __restrict__ cb,     // D_INNER
    float* __restrict__ xc)           // NROWS x D_INNER
{
    const size_t idx = (size_t)blockIdx.x * 256 + threadIdx.x;   // < NROWS*D_INNER
    const int d  = (int)(idx % D_INNER);
    const int bt = (int)(idx / D_INNER);
    const int t  = bt % T_LEN;
    const float* base = xz + (size_t)bt * N_XZ + d;
    float acc = cb[d];
    #pragma unroll
    for (int j = 0; j < D_CONV; ++j) {
        const int tt = t - (D_CONV - 1) + j;
        if (tt >= 0)
            acc += cw[d * D_CONV + j] * base[(ptrdiff_t)(tt - t) * N_XZ];
    }
    xc[idx] = acc / (1.f + __expf(-acc));        // SiLU
}

// ---------------- 5) ssm_params[row][0..32] = xc[row] @ W_x ----------------
__global__ __launch_bounds__(256) void ssm_proj_kernel(
    const float* __restrict__ xc,     // NROWS x D_INNER
    const float* __restrict__ Wx,     // D_INNER x 33
    float* __restrict__ out)          // NROWS x 33
{
    const int row = blockIdx.x;
    float acc[33];
    #pragma unroll
    for (int j = 0; j < 33; ++j) acc[j] = 0.f;
    for (int d = threadIdx.x; d < D_INNER; d += 256) {
        const float v = xc[(size_t)row * D_INNER + d];
        const float* wr = Wx + (size_t)d * 33;
        #pragma unroll
        for (int j = 0; j < 33; ++j) acc[j] += v * wr[j];
    }
    __shared__ float red[256];
    for (int j = 0; j < 33; ++j) {
        red[threadIdx.x] = acc[j];
        __syncthreads();
        for (int s = 128; s > 0; s >>= 1) {
            if (threadIdx.x < s) red[threadIdx.x] += red[threadIdx.x + s];
            __syncthreads();
        }
        if (threadIdx.x == 0) out[(size_t)row * 33 + j] = red[0];
        __syncthreads();
    }
}

// ---------------- 6) sequential SSM scan + gating, emit f16 ----------------
__global__ __launch_bounds__(256) void ssm_scan_kernel(
    const float* __restrict__ params, // NROWS x 33
    const float* __restrict__ xc,     // NROWS x D_INNER
    const float* __restrict__ xz,     // NROWS x N_XZ (z = cols [D_INNER, 2*D_INNER))
    const float* __restrict__ w_dt, const float* __restrict__ b_dt,
    const float* __restrict__ A_log, const float* __restrict__ Dp,
    _Float16* __restrict__ y16)       // NROWS x D_INNER
{
    const int blocks_per_b = D_INNER / 256;      // 8
    const int b  = blockIdx.x / blocks_per_b;
    const int d  = (blockIdx.x % blocks_per_b) * 256 + threadIdx.x;
    float A[D_STATE], h[D_STATE];
    #pragma unroll
    for (int s = 0; s < D_STATE; ++s) {
        A[s] = -__expf(A_log[(size_t)d * D_STATE + s]);
        h[s] = 0.f;
    }
    const float wdt = w_dt[d], bdt = b_dt[d], Dv = Dp[d];
    __shared__ float p[33];

    for (int t = 0; t < T_LEN; ++t) {
        const size_t bt = (size_t)b * T_LEN + t;
        if (threadIdx.x < 33) p[threadIdx.x] = params[bt * 33 + threadIdx.x];
        __syncthreads();
        const float u  = p[0] * wdt + bdt;
        const float dt = (u > 20.f) ? u : log1pf(__expf(u));   // softplus
        const float xv = xc[bt * D_INNER + d];
        float y = 0.f;
        #pragma unroll
        for (int s = 0; s < D_STATE; ++s) {
            const float a = __expf(dt * A[s]);
            h[s] = a * h[s] + dt * p[1 + s] * xv;
            y += h[s] * p[1 + D_STATE + s];
        }
        y += xv * Dv;
        const float z = xz[bt * N_XZ + D_INNER + d];
        y *= z / (1.f + __expf(-z));                           // * silu(z)
        y16[bt * D_INNER + d] = (_Float16)y;
        __syncthreads();
    }
}

// ---------------- workspace layout (bytes) ----------------
static constexpr size_t OFF_XN16   = 0;                                   // 4096x1024 f16
static constexpr size_t OFF_WIN_T  = OFF_XN16  + (size_t)NROWS*D_MODEL*2; // 4096x1024 f16
static constexpr size_t OFF_WOUT_T = OFF_WIN_T + (size_t)N_XZ*D_MODEL*2;  // 1024x2048 f16
static constexpr size_t OFF_XZ     = OFF_WOUT_T+ (size_t)D_MODEL*D_INNER*2;// 4096x4096 f32
static constexpr size_t OFF_XC     = OFF_XZ    + (size_t)NROWS*N_XZ*4;    // 4096x2048 f32
static constexpr size_t OFF_PARAMS = OFF_XC    + (size_t)NROWS*D_INNER*4; // 4096x33 f32
static constexpr size_t OFF_Y16    = OFF_PARAMS+ (size_t)NROWS*33*4 + 128;// 4096x2048 f16

extern "C" void kernel_launch(void* const* d_in, const int* in_sizes, int n_in,
                              void* d_out, int out_size, void* d_ws, size_t ws_size,
                              hipStream_t stream) {
    (void)in_sizes; (void)n_in; (void)out_size; (void)ws_size;
    const float* x      = (const float*)d_in[0];   // (B,T,D_MODEL)
    const float* norm_w = (const float*)d_in[1];
    const float* W_in   = (const float*)d_in[2];   // (D_MODEL, 2*D_INNER)
    const float* conv_w = (const float*)d_in[3];   // (D_INNER,1,4)
    const float* conv_b = (const float*)d_in[4];
    const float* W_x    = (const float*)d_in[5];   // (D_INNER, 33)
    const float* w_dt   = (const float*)d_in[6];
    const float* b_dt   = (const float*)d_in[7];
    const float* A_log  = (const float*)d_in[8];   // (D_INNER, D_STATE)
    const float* D_par  = (const float*)d_in[9];
    const float* W_out  = (const float*)d_in[10];  // (D_INNER, D_MODEL)
    float* out = (float*)d_out;

    char* ws = (char*)d_ws;
    _Float16* xn16   = (_Float16*)(ws + OFF_XN16);
    _Float16* WinT   = (_Float16*)(ws + OFF_WIN_T);
    _Float16* WoutT  = (_Float16*)(ws + OFF_WOUT_T);
    float*    xzbuf  = (float*)   (ws + OFF_XZ);
    float*    xcbuf  = (float*)   (ws + OFF_XC);
    float*    pbuf   = (float*)   (ws + OFF_PARAMS);
    _Float16* y16    = (_Float16*)(ws + OFF_Y16);

    // 1) RMSNorm -> f16
    rmsnorm_cast_kernel<<<NROWS, 256, 0, stream>>>(x, norm_w, xn16);
    // 2) W_in^T, W_out^T -> f16
    transpose_cast_kernel<<<dim3(N_XZ/32, D_MODEL/32), 256, 0, stream>>>(
        W_in, WinT, D_MODEL, N_XZ);
    transpose_cast_kernel<<<dim3(D_MODEL/32, D_INNER/32), 256, 0, stream>>>(
        W_out, WoutT, D_INNER, D_MODEL);
    // 3) xz = xn @ W_in          (M=4096, N=4096, K=1024)
    gemm_wmma_f16<false><<<dim3(N_XZ/BN, NROWS/BM), 256, 0, stream>>>(
        xn16, WinT, xzbuf, nullptr, NROWS, N_XZ, D_MODEL);
    // 4) causal depthwise conv + SiLU
    conv_silu_kernel<<<(NROWS * D_INNER) / 256, 256, 0, stream>>>(
        xzbuf, conv_w, conv_b, xcbuf);
    // 5) ssm_params = xc @ W_x   (N=33)
    ssm_proj_kernel<<<NROWS, 256, 0, stream>>>(xcbuf, W_x, pbuf);
    // 6) sequential scan + D skip + silu(z) gate -> f16
    ssm_scan_kernel<<<B_SZ * (D_INNER / 256), 256, 0, stream>>>(
        pbuf, xcbuf, xzbuf, w_dt, b_dt, A_log, D_par, y16);
    // 7) out = y @ W_out + x     (M=4096, N=1024, K=2048)
    gemm_wmma_f16<true><<<dim3(D_MODEL/BN, NROWS/BM), 256, 0, stream>>>(
        y16, WoutT, out, x, NROWS, D_MODEL, D_INNER);
}
